// GEMM_W4A4_11965778887018
// MI455X (gfx1250) — 8-bit matrix operand order repaired and checked by reading, compiled, not run on hardware
//
#include <hip/hip_runtime.h>
#include <hip/hip_bf16.h>
#include <cstdint>

#define IN_F    3072
#define OUT_F   9216
#define M_TOK   4096
#define NGROUP  48
#define RANK    32

typedef int v8i __attribute__((ext_vector_type(8)));

union FragU {
    v8i       v;
    uint64_t  d[4];
    uint4     q[2];
};

typedef int v4i_gcc __attribute__((vector_size(16)));
typedef __attribute__((address_space(1))) v4i_gcc* as1_v4i_ptr;
typedef __attribute__((address_space(3))) v4i_gcc* as3_v4i_ptr;

#if __has_builtin(__builtin_amdgcn_global_load_async_to_lds_b128)
__device__ __forceinline__ void async_copy16(uint32_t lds_off, const void* gptr) {
    __builtin_amdgcn_global_load_async_to_lds_b128(
        (as1_v4i_ptr)(uintptr_t)gptr,
        (as3_v4i_ptr)(uintptr_t)lds_off,
        0, 0);
}
#else
__device__ __forceinline__ void async_copy16(uint32_t lds_off, const void* gptr) {
    uint64_t ga = (uint64_t)(uintptr_t)gptr;
    asm volatile("global_load_async_to_lds_b128 %0, %1, off"
                 :: "v"(lds_off), "v"(ga) : "memory");
}
#endif

#if __has_builtin(__builtin_amdgcn_s_wait_asynccnt)
#define WAIT_ASYNC_4() __builtin_amdgcn_s_wait_asynccnt(4)
#define WAIT_ASYNC_0() __builtin_amdgcn_s_wait_asynccnt(0)
#else
#define WAIT_ASYNC_4() asm volatile("s_wait_asynccnt 0x4" ::: "memory")
#define WAIT_ASYNC_0() asm volatile("s_wait_asynccnt 0x0" ::: "memory")
#endif

__global__ void unpack_w4(const int8_t* __restrict__ qw, int8_t* __restrict__ w8) {
    int idx = blockIdx.x * 256 + threadIdx.x;
    uint32_t q4 = ((const uint32_t*)qw)[idx];
    uint32_t u[2];
#pragma unroll
    for (int i = 0; i < 2; ++i) {
        int8_t b0 = (int8_t)(q4 >> (16 * i));
        int8_t b1 = (int8_t)(q4 >> (16 * i + 8));
        int8_t l0 = (int8_t)((int8_t)(b0 << 4) >> 4);
        int8_t h0 = (int8_t)(b0 >> 4);
        int8_t l1 = (int8_t)((int8_t)(b1 << 4) >> 4);
        int8_t h1 = (int8_t)(b1 >> 4);
        u[i] = (uint32_t)(uint8_t)l0
             | ((uint32_t)(uint8_t)h0 << 8)
             | ((uint32_t)(uint8_t)l1 << 16)
             | ((uint32_t)(uint8_t)h1 << 24);
    }
    ((uint2*)w8)[idx] = make_uint2(u[0], u[1]);
}

__global__ void quant_act(const float* __restrict__ x, const float* __restrict__ smooth,
                          int8_t* __restrict__ a8, float* __restrict__ ascale_t) {
    const int m    = blockIdx.x;
    const int lane = threadIdx.x & 31;
#pragma unroll
    for (int p = 0; p < 3; ++p) {
        int k = p * 1024 + threadIdx.x * 4;
        float4 xv = *(const float4*)(x + (size_t)m * IN_F + k);
        float4 sv = *(const float4*)(smooth + k);
        float4 xs = make_float4(xv.x / sv.x, xv.y / sv.y, xv.z / sv.z, xv.w / sv.w);
        float amax = fmaxf(fmaxf(fabsf(xs.x), fabsf(xs.y)),
                           fmaxf(fabsf(xs.z), fabsf(xs.w)));
#pragma unroll
        for (int off = 1; off < 16; off <<= 1)
            amax = fmaxf(amax, __shfl_xor(amax, off, 32));
        float scale = fmaxf(amax * (1.0f / 7.0f), 1e-8f);
        float inv   = 1.0f / scale;
        int q0 = (int)fminf(7.f, fmaxf(-8.f, rintf(xs.x * inv)));
        int q1 = (int)fminf(7.f, fmaxf(-8.f, rintf(xs.y * inv)));
        int q2 = (int)fminf(7.f, fmaxf(-8.f, rintf(xs.z * inv)));
        int q3 = (int)fminf(7.f, fmaxf(-8.f, rintf(xs.w * inv)));
        uint32_t packed = (uint32_t)(q0 & 0xFF) | ((uint32_t)(q1 & 0xFF) << 8)
                        | ((uint32_t)(q2 & 0xFF) << 16) | ((uint32_t)(q3 & 0xFF) << 24);
        *(uint32_t*)(a8 + (size_t)m * IN_F + k) = packed;
        if ((lane & 15) == 0)
            ascale_t[(size_t)(k >> 6) * M_TOK + m] = scale;
    }
}

__global__ void lora_down_act(const float* __restrict__ x, const float* __restrict__ smooth,
                              const float* __restrict__ ld, float* __restrict__ lora_act) {
    __shared__ __align__(16) float xsl[16][128];
    __shared__ __align__(16) float ldl[128][RANK];
    const int t  = threadIdx.x;
    const int m0 = blockIdx.x * 16;
    const int r  = t & 31;
    const int rh = t >> 5;
    float p0 = 0.f, p1 = 0.f;
    for (int c = 0; c < IN_F / 128; ++c) {
        __syncthreads();
#pragma unroll
        for (int j = 0; j < 2; ++j) {
            int lin = t * 2 + j;
            int row = lin >> 5, kk = (lin & 31) * 4;
            float4 xv = *(const float4*)(x + (size_t)(m0 + row) * IN_F + c * 128 + kk);
            float4 sv = *(const float4*)(smooth + c * 128 + kk);
            *(float4*)&xsl[row][kk] =
                make_float4(xv.x / sv.x, xv.y / sv.y, xv.z / sv.z, xv.w / sv.w);
        }
#pragma unroll
        for (int j = 0; j < 4; ++j) {
            int off = (t * 4 + j) * 4;
            *(float4*)&(((float*)ldl)[off]) =
                *(const float4*)(ld + (size_t)c * 128 * RANK + off);
        }
        __syncthreads();
        for (int kk = 0; kk < 128; ++kk) {
            float lv = ldl[kk][r];
            p0 += xsl[rh][kk]     * lv;
            p1 += xsl[rh + 8][kk] * lv;
        }
    }
    lora_act[(size_t)(m0 + rh) * RANK + r]     = p0;
    lora_act[(size_t)(m0 + rh + 8) * RANK + r] = p1;
}

__global__ void lora_up_bias(const float* __restrict__ lora_act, const float* __restrict__ lu,
                             const float* __restrict__ bias, float* __restrict__ out) {
    __shared__ __align__(16) float la[16][RANK];
    __shared__ __align__(16) float luS[64][RANK];
    const int t  = threadIdx.x;
    const int n0 = blockIdx.x * 64;
    const int m0 = blockIdx.y * 16;
    if (t < 128) {
        int row = t >> 3, rr = (t & 7) * 4;
        *(float4*)&la[row][rr] = *(const float4*)(lora_act + (size_t)(m0 + row) * RANK + rr);
    }
#pragma unroll
    for (int j = 0; j < 2; ++j) {
        int lin = t * 2 + j;
        int n = lin >> 3, rr = (lin & 7) * 4;
        *(float4*)&luS[n][rr] = *(const float4*)(lu + (size_t)(n0 + n) * RANK + rr);
    }
    __syncthreads();
    const int row = t >> 4;
    const int c4  = (t & 15) * 4;
    float acc[4];
#pragma unroll
    for (int i = 0; i < 4; ++i) acc[i] = bias[n0 + c4 + i];
    for (int r = 0; r < RANK; ++r) {
        float a = la[row][r];
#pragma unroll
        for (int i = 0; i < 4; ++i) acc[i] += a * luS[c4 + i][r];
    }
    *(float4*)(out + (size_t)(m0 + row) * OUT_F + n0 + c4) =
        make_float4(acc[0], acc[1], acc[2], acc[3]);
}

extern __shared__ __align__(16) int8_t smem[];

__device__ __forceinline__ void stage_tiles_async(const int8_t* gA, const int8_t* gB,
                                                  uint32_t baseA, uint32_t baseB, int t) {
#pragma unroll
    for (int it = 0; it < 2; ++it) {
        int idx = t + it * 256;
        int row = idx >> 2, off = (idx & 3) * 16;
        async_copy16(baseA + row * 64 + off, gA + (size_t)row * IN_F + off);
        async_copy16(baseB + row * 64 + off, gB + (size_t)row * IN_F + off);
    }
}

__global__ __launch_bounds__(256)
void gemm_w4a4(const int8_t* __restrict__ a8, const int8_t* __restrict__ w8,
               const float* __restrict__ ascale_t, const float* __restrict__ wscales,
               float* __restrict__ out) {
    const int t     = threadIdx.x;
    const int lane  = t & 31;
    const int wave  = t >> 5;
    const int wm    = wave & 3;
    const int wn    = wave >> 2;
    const int m0    = blockIdx.y * 128;
    const int n0    = blockIdx.x * 128;
    const int khalf = lane >> 4;
    const int ln    = lane & 15;
    const int rowbase = m0 + wm * 32;
    const int colbase = n0 + wn * 64;

    const int8_t* gA0 = a8 + (size_t)m0 * IN_F;
    const int8_t* gB0 = w8 + (size_t)n0 * IN_F;

    float acc[2][4][8];
#pragma unroll
    for (int tm = 0; tm < 2; ++tm)
#pragma unroll
        for (int tn = 0; tn < 4; ++tn)
#pragma unroll
            for (int v = 0; v < 8; ++v) {
                int r = rowbase + tm * 16 + khalf * 8 + v;
                int c = colbase + tn * 16 + ln;
                acc[tm][tn][v] = out[(size_t)r * OUT_F + c];
            }

    stage_tiles_async(gA0, gB0, 0u, 8192u, t);

    for (int kg = 0; kg < NGROUP; ++kg) {
        const int nxt = kg + 1;
        if (nxt < NGROUP) {
            uint32_t pb = (uint32_t)(nxt & 1) * 16384u;
            stage_tiles_async(gA0 + (size_t)nxt * 64, gB0 + (size_t)nxt * 64,
                              pb, pb + 8192u, t);
            if (kg + 2 < NGROUP) {
                int idx = t, row = idx >> 2, off = (idx & 3) * 16;
                __builtin_prefetch(gA0 + (size_t)(kg + 2) * 64 + (size_t)row * IN_F + off, 0, 1);
                __builtin_prefetch(gB0 + (size_t)(kg + 2) * 64 + (size_t)row * IN_F + off, 0, 1);
            }
            WAIT_ASYNC_4();
        } else {
            WAIT_ASYNC_0();
        }
        __syncthreads();

        const uint32_t cb = (uint32_t)(kg & 1) * 16384u;
        const int8_t* ldsA = smem + cb;
        const int8_t* ldsB = smem + cb + 8192u;

        FragU A[2];
#pragma unroll
        for (int tm = 0; tm < 2; ++tm) {
            const int8_t* pa = &ldsA[(wm * 32 + tm * 16 + ln) * 64 + khalf * 16];
            A[tm].d[0] = *(const uint64_t*)(pa + 0);
            A[tm].d[1] = *(const uint64_t*)(pa + 8);
            A[tm].d[2] = *(const uint64_t*)(pa + 32);
            A[tm].d[3] = *(const uint64_t*)(pa + 40);
        }
        FragU B[4];
#pragma unroll
        for (int tn = 0; tn < 4; ++tn) {
            const int8_t* pb = &ldsB[(wn * 64 + tn * 16 + ln) * 64 + khalf * 16];
            B[tn].q[0] = *(const uint4*)(pb + 0);
            B[tn].q[1] = *(const uint4*)(pb + 32);
        }
        float as_[2][8];
#pragma unroll
        for (int tm = 0; tm < 2; ++tm) {
            const float* asp = ascale_t + (size_t)kg * M_TOK + rowbase + tm * 16 + khalf * 8;
            float4 lo = *(const float4*)asp;
            float4 hi = *(const float4*)(asp + 4);
            as_[tm][0] = lo.x; as_[tm][1] = lo.y; as_[tm][2] = lo.z; as_[tm][3] = lo.w;
            as_[tm][4] = hi.x; as_[tm][5] = hi.y; as_[tm][6] = hi.z; as_[tm][7] = hi.w;
        }
        float ws_[4];
#pragma unroll
        for (int tn = 0; tn < 4; ++tn)
            ws_[tn] = wscales[(size_t)kg * OUT_F + colbase + tn * 16 + ln];

#pragma unroll
        for (int tm = 0; tm < 2; ++tm)
#pragma unroll
            for (int tn = 0; tn < 4; ++tn) {
                v8i c = {0, 0, 0, 0, 0, 0, 0, 0};
                c = __builtin_amdgcn_wmma_i32_16x16x64_iu8(
                        true, A[tm].v, true, B[tn].v, c, false, false);
#pragma unroll
                for (int v = 0; v < 8; ++v)
                    acc[tm][tn][v] += (float)c[v] * (as_[tm][v] * ws_[tn]);
            }
        __syncthreads();
    }

#pragma unroll
    for (int tm = 0; tm < 2; ++tm)
#pragma unroll
        for (int tn = 0; tn < 4; ++tn)
#pragma unroll
            for (int v = 0; v < 8; ++v) {
                int r = rowbase + tm * 16 + khalf * 8 + v;
                int c = colbase + tn * 16 + ln;
                out[(size_t)r * OUT_F + c] = acc[tm][tn][v];
            }
}

__global__ void norm_rope(float* __restrict__ out, const float* __restrict__ nq,
                          const float* __restrict__ nk, const float* __restrict__ rot) {
    const int lane = threadIdx.x & 31;
    const int wave = threadIdx.x >> 5;
    int row  = blockIdx.x * 8 + wave;
    int m    = row / 48;
    int h    = row % 48;
    int isK  = (h >= 24) ? 1 : 0;
    int head = isK ? (h - 24) : h;
    size_t base = (size_t)m * OUT_F + (size_t)isK * 3072 + (size_t)head * 128 + lane * 4;

    float4 tv = *(float4*)(out + base);
    float ss = tv.x * tv.x + tv.y * tv.y + tv.z * tv.z + tv.w * tv.w;
#pragma unroll
    for (int off = 1; off < 32; off <<= 1)
        ss += __shfl_xor(ss, off, 32);
    float rn = rsqrtf(ss * (1.0f / 128.0f) + 1e-6f);

    const float* g = isK ? nk : nq;
    float4 gv = *(const float4*)(g + lane * 4);
    tv.x *= rn * gv.x; tv.y *= rn * gv.y; tv.z *= rn * gv.z; tv.w *= rn * gv.w;

    float4 cs = *(const float4*)(rot + (size_t)m * 128 + lane * 4);
    float e0 = tv.x, o0 = tv.y, e1 = tv.z, o1 = tv.w;
    tv.x = e0 * cs.x - o0 * cs.y;
    tv.y = e0 * cs.y + o0 * cs.x;
    tv.z = e1 * cs.z - o1 * cs.w;
    tv.w = e1 * cs.w + o1 * cs.z;
    *(float4*)(out + base) = tv;
}

extern "C" void kernel_launch(void* const* d_in, const int* in_sizes, int n_in,
                              void* d_out, int out_size, void* d_ws, size_t ws_size,
                              hipStream_t stream) {
    const float*  x         = (const float*)d_in[0];
    const int8_t* qweight   = (const int8_t*)d_in[1];
    const float*  wscales   = (const float*)d_in[2];
    const float*  bias      = (const float*)d_in[3];
    const float*  lora_down = (const float*)d_in[4];
    const float*  lora_up   = (const float*)d_in[5];
    const float*  smooth    = (const float*)d_in[6];
    const float*  norm_q    = (const float*)d_in[7];
    const float*  norm_k    = (const float*)d_in[8];
    const float*  rot       = (const float*)d_in[9];
    float* out = (float*)d_out;

    int8_t* w8       = (int8_t*)d_ws;
    int8_t* a8       = w8 + (size_t)OUT_F * IN_F;
    float*  ascale_t = (float*)(a8 + (size_t)M_TOK * IN_F);
    float*  lora_act = ascale_t + (size_t)NGROUP * M_TOK;

    unpack_w4<<<(OUT_F * IN_F / 2 / 4) / 256, 256, 0, stream>>>(qweight, w8);
    quant_act<<<M_TOK, 256, 0, stream>>>(x, smooth, a8, ascale_t);
    lora_down_act<<<M_TOK / 16, 256, 0, stream>>>(x, smooth, lora_down, lora_act);
    lora_up_bias<<<dim3(OUT_F / 64, M_TOK / 16), 256, 0, stream>>>(lora_act, lora_up, bias, out);
    gemm_w4a4<<<dim3(OUT_F / 128, M_TOK / 128), 256, 32768, stream>>>(a8, w8, ascale_t, wscales, out);
    norm_rope<<<M_TOK * 48 / 8, 256, 0, stream>>>(out, norm_q, norm_k, rot);
}
